// WaveNet_2001454760676
// MI455X (gfx1250) — matrix operand order repaired and checked by reading, compiled, not run on hardware
//
#include <hip/hip_runtime.h>

typedef __attribute__((ext_vector_type(16))) _Float16 v16h;
typedef __attribute__((ext_vector_type(8)))  float    v8f;

#define CH     64
#define TLEN   32768
#define BATCH  4
#define LAYERS 18
#define TILES  4                       // 16-step tiles per wave per block

__device__ __forceinline__ float fast_sigmoid(float x) {
    return 1.f / (1.f + __expf(-x));
}
__device__ __forceinline__ float fast_tanh(float x) {
    // robust at +-inf: 1 - 2/(e^{2x}+1)
    float e = __expf(2.f * x);
    return 1.f - 2.f / (e + 1.f);
}

// ---------------------------------------------------------------------------
// input 1x1 conv: act[b,c,t] = in_w[c]*x[b,t] + in_b[c]
// ---------------------------------------------------------------------------
__global__ void k_input(const float* __restrict__ x,
                        const float* __restrict__ in_w,
                        const float* __restrict__ in_b,
                        float* __restrict__ act) {
    int i = blockIdx.x * 256 + threadIdx.x;          // over B*C*T
    int t  = i % TLEN;
    int bc = i / TLEN;
    int c  = bc % CH;
    int b  = bc / CH;
    act[i] = in_w[c] * x[b * TLEN + t] + in_b[c];
}

__global__ void k_zero(float* __restrict__ p) {
    p[blockIdx.x * 256 + threadIdx.x] = 0.f;
}

__global__ void k_final(const float* __restrict__ skip,
                        const float* __restrict__ mix_b,
                        float* __restrict__ out) {
    int i = blockIdx.x * 256 + threadIdx.x;          // over B*T
    out[i] = skip[i] + mix_b[0];
}

// ---------------------------------------------------------------------------
// One WaveNet layer, fully fused:
//   H = Wh(conv, dil) * inp + bh          (128 x 16 per time-tile, WMMA f16)
//   G = tanh(H[:64]) * sigmoid(H[64:])
//   skip += mix_w_l . G                   (folded mix conv)
//   out = Wr * G + br + inp               (WMMA f16)
// Block = 256 threads = 8 waves; each wave owns TILES consecutive 16-step
// tiles (amortizes the once-per-block LDS weight staging 4x).
// All per-layer weights are staged pre-swizzled into WMMA A-fragment order
// so each lane reads contiguous 32B via ds_load_b128 pairs. A memory
// barrier per tile iteration keeps fragment reads in-loop (prevents the
// compiler from hoisting 48 fragments into VGPRs and spilling to scratch).
// ---------------------------------------------------------------------------
__global__ __launch_bounds__(256)
void k_layer(const float* __restrict__ inp, float* __restrict__ outp,
             float* __restrict__ skip,
             const float* __restrict__ hid_w, const float* __restrict__ hid_b,
             const float* __restrict__ res_w, const float* __restrict__ res_b,
             const float* __restrict__ mix_w,
             int layer, int dil) {
    // hid weights in A-fragment order: 48 frags x 32 lanes x 16 halves (48 KB)
    __shared__ __align__(32) _Float16 sm_hid[48 * 512];
    // res weights in A-fragment order: 8 frags x 32 lanes x 16 halves (8 KB)
    __shared__ __align__(32) _Float16 sm_res[8 * 512];
    // gated activations in B-fragment order: [wave][kb][lane][16] (16 KB)
    __shared__ __align__(32) _Float16 sm_g[8][2][32][16];
    // small per-layer vectors (~1 KB)
    __shared__ float sm_hb[128];
    __shared__ float sm_rb[64];
    __shared__ float sm_mw[64];

    const int tid = threadIdx.x;

    // ---- stage hid_w (f32 -> f16, pre-swizzled into A-fragment layout) ----
    for (int idx = tid; idx < 48 * 512; idx += 256) {
        int frag = idx >> 9;
        int rem9 = idx & 511;
        int ln   = rem9 >> 4;
        int e    = rem9 & 15;
        int mt = frag / 6;
        int r6 = frag % 6;
        int s  = r6 >> 1;        // tap: x offset = s*dil, weight j = 2-s
        int kb = r6 & 1;
        int m  = mt * 16 + (ln & 15);
        int hw = ln >> 4;
        int kl = hw * 16 + e;   // contiguous k = 16*hw + e (B operand uses the same order)
        int k  = kb * 32 + kl;
        int j  = 2 - s;
        sm_hid[idx] = (_Float16)hid_w[(((size_t)layer * 128 + m) * 64 + k) * 3 + j];
    }
    // ---- stage res_w the same way ----
    for (int idx = tid; idx < 8 * 512; idx += 256) {
        int frag = idx >> 9;
        int rem9 = idx & 511;
        int ln   = rem9 >> 4;
        int e    = rem9 & 15;
        int mt = frag >> 1;
        int kb = frag & 1;
        int m  = mt * 16 + (ln & 15);
        int hw = ln >> 4;
        int kl = hw * 16 + e;
        int k  = kb * 32 + kl;
        sm_res[idx] = (_Float16)res_w[((size_t)layer * 64 + m) * 64 + k];
    }
    // ---- stage per-layer vectors ----
    if (tid < 128) sm_hb[tid] = hid_b[layer * 128 + tid];
    if (tid < 64)  sm_rb[tid] = res_b[layer * 64 + tid];
    if (tid >= 64 && tid < 128) sm_mw[tid - 64] = mix_w[layer * 64 + (tid - 64)];
    __syncthreads();

    const int lane = tid & 31;
    const int wave = tid >> 5;
    const int n  = lane & 15;     // column (timestep within tile)
    const int hi = lane >> 4;     // half-wave

    const int blocksPerB = TLEN / (128 * TILES);
    const int b    = blockIdx.x / blocksPerB;
    const int base = (blockIdx.x % blocksPerB) * (128 * TILES);
    const float* __restrict__ inB  = inp  + (size_t)b * CH * TLEN;
    float*       __restrict__ outB = outp + (size_t)b * CH * TLEN;

#pragma unroll 1
    for (int it = 0; it < TILES; ++it) {
        // keep LDS fragment reads inside the loop (see header comment)
        asm volatile("" ::: "memory");

        const int t0 = base + (wave * TILES + it) * 16;

        // -- B fragments for the 3 causal taps (t, t-d, t-2d), causal pad ---
        v16h xb[3][2];
        for (int s = 0; s < 3; ++s) {
            int t = t0 + n - s * dil;
            for (int kb = 0; kb < 2; ++kb) {
                v16h v;
                for (int j = 0; j < 16; ++j) {
                    int c = kb * 32 + hi * 16 + j;    // contiguous k = 16*hi + j (matches A staging order)
                    float f = (t >= 0) ? inB[(size_t)c * TLEN + t] : 0.f;
                    v[j] = (_Float16)f;
                }
                xb[s][kb] = v;
            }
        }

        // -- hid GEMM: 8 M-tiles x (3 taps x 2 K-blocks) = 48 WMMAs ---------
        v8f acc[8];
        for (int mt = 0; mt < 8; ++mt) {
            v8f a = {};
            for (int s = 0; s < 3; ++s)
                for (int kb = 0; kb < 2; ++kb) {
                    const v16h wf =
                        *(const v16h*)&sm_hid[((mt * 6 + s * 2 + kb) << 9) + (lane << 4)];
                    a = __builtin_amdgcn_wmma_f32_16x16x32_f16(
                            false, wf, false, xb[s][kb], (short)0, a, false, false);
                }
            acc[mt] = a;
        }

        // -- bias + gating; scatter G into LDS in B-fragment order ----------
        for (int mt = 0; mt < 4; ++mt) {
            for (int r = 0; r < 8; ++r) {
                int m = mt * 16 + hi * 8 + r;         // gated channel 0..63
                float hA = acc[mt][r]     + sm_hb[m];
                float hG = acc[mt + 4][r] + sm_hb[64 + m];
                float g  = fast_tanh(hA) * fast_sigmoid(hG);
                int kb  = m >> 5;
                int km  = m & 31;
                int hib = km >> 4;
                int jj  = km & 15;
                sm_g[wave][kb][hib * 16 + n][jj] = (_Float16)g;
            }
        }

        // -- read G back (contiguous 32B per lane): res B op + skip mix -----
        v16h gb0 = *(const v16h*)&sm_g[wave][0][lane][0];
        v16h gb1 = *(const v16h*)&sm_g[wave][1][lane][0];

        float part = 0.f;
        for (int j = 0; j < 16; ++j) {
            part += (float)gb0[j] * sm_mw[     hi * 16 + j];
            part += (float)gb1[j] * sm_mw[32 + hi * 16 + j];
        }
        float oth = __shfl_down(part, 16, 32);
        if (hi == 0) {
            int t = t0 + n;                           // exclusive owner of (b,t)
            skip[(size_t)b * TLEN + t] += part + oth;
        }
        // (reconverged: EXEC all ones again before the WMMAs below)

        // -- res conv: 4 M-tiles x 2 K-blocks = 8 WMMAs ---------------------
        v8f racc[4];
        for (int mt = 0; mt < 4; ++mt) {
            v8f a = {};
            for (int kb = 0; kb < 2; ++kb) {
                const v16h wf =
                    *(const v16h*)&sm_res[((mt * 2 + kb) << 9) + (lane << 4)];
                a = __builtin_amdgcn_wmma_f32_16x16x32_f16(
                        false, wf, false, (kb ? gb1 : gb0), (short)0, a, false, false);
            }
            racc[mt] = a;
        }

        // -- out = res + bias + residual; store -----------------------------
        for (int mt = 0; mt < 4; ++mt) {
            for (int r = 0; r < 8; ++r) {
                int m = mt * 16 + hi * 8 + r;
                int t = t0 + n;
                float v = racc[mt][r] + sm_rb[m] + inB[(size_t)m * TLEN + t];
                outB[(size_t)m * TLEN + t] = v;
            }
        }
    }
}

// ---------------------------------------------------------------------------
extern "C" void kernel_launch(void* const* d_in, const int* in_sizes, int n_in,
                              void* d_out, int out_size, void* d_ws, size_t ws_size,
                              hipStream_t stream) {
    (void)in_sizes; (void)n_in; (void)out_size; (void)ws_size;
    const float* x     = (const float*)d_in[0];
    const float* in_w  = (const float*)d_in[1];
    const float* in_b  = (const float*)d_in[2];
    const float* hid_w = (const float*)d_in[3];
    const float* hid_b = (const float*)d_in[4];
    const float* res_w = (const float*)d_in[5];
    const float* res_b = (const float*)d_in[6];
    const float* mix_w = (const float*)d_in[7];
    const float* mix_b = (const float*)d_in[8];
    float* out = (float*)d_out;

    const size_t actElems = (size_t)BATCH * CH * TLEN;   // 8,388,608 floats
    float* act0 = (float*)d_ws;
    float* act1 = act0 + actElems;
    float* skip = act1 + actElems;                        // BATCH*TLEN floats

    k_input<<<(BATCH * CH * TLEN) / 256, 256, 0, stream>>>(x, in_w, in_b, act0);
    k_zero<<<(BATCH * TLEN) / 256, 256, 0, stream>>>(skip);

    const float* cur = act0;
    float* nxt = act1;
    for (int l = 0; l < LAYERS; ++l) {
        int dil = 1 << (l % 9);
        k_layer<<<BATCH * (TLEN / (128 * TILES)), 256, 0, stream>>>(
            cur, nxt, skip, hid_w, hid_b, res_w, res_b, mix_w, l, dil);
        const float* tmp = nxt;
        nxt = (float*)cur;
        cur = tmp;
    }

    k_final<<<(BATCH * TLEN) / 256, 256, 0, stream>>>(skip, mix_b, out);
}
